// GraphConv_AllFeats_21474836480770
// MI455X (gfx1250) — matrix operand order repaired and checked by reading, not compiled, not run on hardware
//
#include <hip/hip_runtime.h>
#include <hip/hip_bf16.h>
#include <math.h>

typedef __attribute__((ext_vector_type(16))) _Float16 v16h;
typedef __attribute__((ext_vector_type(8)))  _Float16 v8h;
typedef __attribute__((ext_vector_type(8)))  float    v8f;

// ---------------------------------------------------------------------------
// Zero-fill (agg cleared before each layer's scatter-add)
// ---------------------------------------------------------------------------
__global__ void zero_kernel(float* __restrict__ p, int n) {
    int i = blockIdx.x * blockDim.x + threadIdx.x;
    if (i < n) p[i] = 0.0f;
}

// ---------------------------------------------------------------------------
// Layer-0 input pad: x (N x 13 f32) -> h (N x 64 f16, zero-padded K)
// ---------------------------------------------------------------------------
__global__ void pad_x_kernel(const float* __restrict__ x,
                             _Float16* __restrict__ h, int N) {
    int i = blockIdx.x * blockDim.x + threadIdx.x;
    if (i >= N * 64) return;
    int row = i >> 6;
    int f   = i & 63;
    h[i] = (_Float16)((f < 13) ? x[row * 13 + f] : 0.0f);
}

// ---------------------------------------------------------------------------
// Weight/bias prep: pack Wr/Wq of every layer into f16 WMMA B-fragments in
// register order [matrix][tile][kstep][lane][16 halves], zero-padded.
// B 32x16 f16 layout (ISA 7.12.2): lane n holds column n; half j holds
// K = 8*hi + j (j<8), 16 + 8*hi + (j-8) (j>=8)  (hi = lane>>4).
// ---------------------------------------------------------------------------
struct PrepParams {
    const float* W[10];   // W[2l]=Wr_l, W[2l+1]=Wq_l
    const float* br[5];
};

__global__ void prep_kernel(PrepParams p, _Float16* __restrict__ frag,
                            float* __restrict__ brpad) {
    const int lane = threadIdx.x;           // 32 threads
    const int id   = blockIdx.x;
    if (id < 80) {
        const int l   = id >> 4;
        const int rem = id & 15;
        const int mtx = rem >> 3;           // 0=Wr, 1=Wq
        const int t   = (rem >> 1) & 3;     // column tile
        const int s   = rem & 1;            // k-step (K base = 32*s)
        const int din  = (l == 0) ? 13 : 64;
        const int dout = (l == 4) ? 1 : 64;
        const float* W = p.W[2 * l + mtx];
        const int hi = lane >> 4;
        const int n  = lane & 15;
        const int col = t * 16 + n;
        _Float16* dst = frag +
            (size_t)((((2 * l + mtx) * 4 + t) * 2 + s) * 32 + lane) * 16;
#pragma unroll
        for (int j = 0; j < 16; ++j) {
            int kk = s * 32 + ((j < 8) ? (8 * hi + j) : (16 + 8 * hi + (j - 8)));
            float v = (kk < din && col < dout) ? W[kk * dout + col] : 0.0f;
            dst[j] = (_Float16)v;
        }
    } else {
        const int l = id - 80;
        const int dout = (l == 4) ? 1 : 64;
        int c0 = lane, c1 = lane + 32;
        brpad[l * 64 + c0] = (c0 < dout) ? p.br[l][c0] : 0.0f;
        brpad[l * 64 + c1] = (c1 < dout) ? p.br[l][c1] : 0.0f;
    }
}

// ---------------------------------------------------------------------------
// Edge scatter: agg[dst] += h[src] * ew   (8-feature b128 chunks, f32 atomics
// into L2 — whole working set is L2-resident on MI455X's 192MB L2)
// ---------------------------------------------------------------------------
__global__ void scatter_kernel(const _Float16* __restrict__ h,
                               const long long* __restrict__ src,
                               const long long* __restrict__ dst,
                               const float* __restrict__ ew,
                               float* __restrict__ agg,
                               int E, int nchunks) {
    int idx = blockIdx.x * blockDim.x + threadIdx.x;
    if (idx >= E * nchunks) return;
    int e  = idx / nchunks;
    int f0 = (idx - e * nchunks) * 8;
    long long s = src[e];
    long long d = dst[e];
    float w = ew[e];
    const v8h hv = *(const v8h*)(h + (size_t)s * 64 + f0);
    float* ar = agg + (size_t)d * 64 + f0;
#pragma unroll
    for (int j = 0; j < 8; ++j)
        atomicAdd(&ar[j], (float)hv[j] * w);
}

// ---------------------------------------------------------------------------
// Fused GEMM: out = act( agg@Wr + br + h@Wq ), K padded to 64.
// One wave per 16x16 tile; branch-free b128 fragment loads; two
// v_wmma_f32_16x16x32_f16 per K-step accumulating into one f32 fragment.
// act: 0 = ReLU -> f16 h buffer (N x 64); 1 = sigmoid -> f32 out (N x dout)
// ---------------------------------------------------------------------------
__global__ void gemm_fused_kernel(const _Float16* __restrict__ h,
                                  const float* __restrict__ agg,
                                  const _Float16* __restrict__ bfragR,
                                  const _Float16* __restrict__ bfragQ,
                                  const float* __restrict__ brpad,
                                  void* __restrict__ out,
                                  int N, int dout, int act) {
    const int lane = threadIdx.x & 31;
    const int wave = threadIdx.x >> 5;      // column tile
    const int hi   = lane >> 4;
    const int m    = lane & 15;
    const int rowBase = blockIdx.x * 16;
    int row = rowBase + m;
    if (row >= N) row = N - 1;              // branch-free clamp (stores guarded)

    const _Float16* hrow = h   + (size_t)row * 64;
    const float*    arow = agg + (size_t)row * 64;

    v8f c = {};
#pragma unroll
    for (int s = 0; s < 2; ++s) {
        const int k0 = s * 32;
        // A from h (f16): two contiguous 8-half (16B) chunks per ISA layout
        v16h aH;
        {
            v8h lo = *(const v8h*)(hrow + k0 + 8 * hi);
            v8h up = *(const v8h*)(hrow + k0 + 16 + 8 * hi);
#pragma unroll
            for (int j = 0; j < 8; ++j) { aH[j] = lo[j]; aH[8 + j] = up[j]; }
        }
        // A from agg (f32): four float4 loads + cvt
        v16h aA;
        {
            const float* p0 = arow + k0 + 8 * hi;
            const float* p1 = arow + k0 + 16 + 8 * hi;
            float4 f0 = *(const float4*)(p0);
            float4 f1 = *(const float4*)(p0 + 4);
            float4 f2 = *(const float4*)(p1);
            float4 f3 = *(const float4*)(p1 + 4);
            aA[0]  = (_Float16)f0.x; aA[1]  = (_Float16)f0.y;
            aA[2]  = (_Float16)f0.z; aA[3]  = (_Float16)f0.w;
            aA[4]  = (_Float16)f1.x; aA[5]  = (_Float16)f1.y;
            aA[6]  = (_Float16)f1.z; aA[7]  = (_Float16)f1.w;
            aA[8]  = (_Float16)f2.x; aA[9]  = (_Float16)f2.y;
            aA[10] = (_Float16)f2.z; aA[11] = (_Float16)f2.w;
            aA[12] = (_Float16)f3.x; aA[13] = (_Float16)f3.y;
            aA[14] = (_Float16)f3.z; aA[15] = (_Float16)f3.w;
        }
        // B fragments: pre-packed, one contiguous 32B load each
        const size_t boff = (size_t)(((wave * 2 + s) * 32) + lane) * 16;
        v16h bR = *(const v16h*)(bfragR + boff);
        v16h bQ = *(const v16h*)(bfragQ + boff);

        c = __builtin_amdgcn_wmma_f32_16x16x32_f16(false, aH, false, bQ,
                                                   (short)0, c, false, false);
        c = __builtin_amdgcn_wmma_f32_16x16x32_f16(false, aA, false, bR,
                                                   (short)0, c, false, false);
    }

    // C/D layout: VGPR r, lane L -> row = r + 8*(L>>4), col = L&15
    const int col = wave * 16 + (lane & 15);
    const float bias = brpad[col & 63];
#pragma unroll
    for (int r = 0; r < 8; ++r) {
        int rr = rowBase + r + 8 * hi;
        float v = c[r] + bias;
        if (act) {
            v = 1.0f / (1.0f + __expf(-v));
            if (rr < N && col < dout)
                ((float*)out)[(size_t)rr * dout + col] = v;
        } else {
            v = fmaxf(v, 0.0f);
            if (rr < N)
                ((_Float16*)out)[(size_t)rr * 64 + col] = (_Float16)v;
        }
    }
}

// ---------------------------------------------------------------------------
// Host-side orchestration
// ---------------------------------------------------------------------------
extern "C" void kernel_launch(void* const* d_in, const int* in_sizes, int n_in,
                              void* d_out, int out_size, void* d_ws, size_t ws_size,
                              hipStream_t stream) {
    const float*     x  = (const float*)d_in[0];
    const long long* ei = (const long long*)d_in[1];   // jnp.int64 edge_index [2, E]
    const float*     ew = (const float*)d_in[2];

    const int N = in_sizes[0] / 13;    // 100000 (multiple of 16)
    const int E = in_sizes[2];         // 1200000

    const long long* src = ei;
    const long long* dst = ei + E;

    // Workspace layout
    float*    agg  = (float*)d_ws;                                  // N*64 f32
    _Float16* hA   = (_Float16*)((char*)d_ws + (size_t)N * 64 * 4); // N*64 f16
    _Float16* hB   = hA + (size_t)N * 64;                           // N*64 f16
    _Float16* frag = hB + (size_t)N * 64;                           // 10*4096 f16
    float*    brp  = (float*)(frag + 10 * 4096);                    // 5*64 f32

    // One-shot prep: pad x to f16, pack all weights into B-fragments
    pad_x_kernel<<<(N * 64 + 255) / 256, 256, 0, stream>>>(x, hA, N);
    {
        PrepParams p;
        for (int l = 0; l < 5; ++l) {
            p.W[2 * l]     = (const float*)d_in[3 + 3 * l];  // Wr_l
            p.W[2 * l + 1] = (const float*)d_in[5 + 3 * l];  // Wq_l
            p.br[l]        = (const float*)d_in[4 + 3 * l];  // br_l
        }
        prep_kernel<<<85, 32, 0, stream>>>(p, frag, brp);
    }

    const _Float16* hcur = hA;
    for (int l = 0; l < 5; ++l) {
        const int dout = (l == 4) ? 1 : 64;
        const int act  = (l == 4) ? 1 : 0;

        // 1) agg = 0
        zero_kernel<<<(N * 64 + 255) / 256, 256, 0, stream>>>(agg, N * 64);

        // 2) agg[dst] += hcur[src] * ew  (layer 0 only needs first 16 cols)
        {
            int nchunks = (l == 0) ? 2 : 8;
            long long total = (long long)E * nchunks;
            scatter_kernel<<<(int)((total + 255) / 256), 256, 0, stream>>>(
                hcur, src, dst, ew, agg, E, nchunks);
        }

        // 3) fused GEMM + bias + activation
        {
            const _Float16* bfR = frag + (size_t)(2 * l)     * 4096;
            const _Float16* bfQ = frag + (size_t)(2 * l + 1) * 4096;
            const float*    bp  = brp + l * 64;
            void* o = (l == 4) ? d_out : (void*)((l & 1) ? hA : hB);
            int ntiles = (dout + 15) / 16;       // 4 waves (dout=64) or 1 (dout=1)
            gemm_fused_kernel<<<(N + 15) / 16, 32 * ntiles, 0, stream>>>(
                hcur, agg, bfR, bfQ, bp, o, N, dout, act);
            if (l < 4) hcur = (const _Float16*)o;
        }
    }
}
